// Sep12LSTMsOneMLP_8263517078069
// MI455X (gfx1250) — compile-verified, matrix operand order checked by reading, not run on hardware
//
#include <hip/hip_runtime.h>
#include <hip/hip_bf16.h>
#include <math.h>

// ---------------- problem constants ----------------
constexpr int Bn  = 4096;
constexpr int Tn  = 24;
constexpr int Fn  = 12;
constexpr int Hn  = 512;
constexpr int GH  = 4 * Hn;          // 2048 gate rows
constexpr int SD  = 64;              // static dim
constexpr int SE  = 256;             // static emb
constexpr int MH  = 1024;            // mlp hidden
constexpr int TGT = 2;
constexpr int ZK  = Fn * Hn + SE;    // 6400
constexpr int HSTR = Hn + 8;         // padded LDS row stride (ushorts)
constexpr int GSTRIDE = 32 * 16 * 512;   // ushort stride between gate blocks in Wp

// ---------------- WMMA fragment types ----------------
typedef __attribute__((ext_vector_type(16))) __bf16 v16bf;
typedef __attribute__((ext_vector_type(8)))  float  v8f;

union Frag { uint4 u[2]; v16bf v; };

__device__ __forceinline__ unsigned short f2bf(float x) {
  unsigned int u = __float_as_uint(x);
  u += 0x7FFFu + ((u >> 16) & 1u);      // round-to-nearest-even
  return (unsigned short)(u >> 16);
}
__device__ __forceinline__ float sigf(float x) { return 1.0f / (1.0f + __expf(-x)); }

// ---------------- weight packing into B-fragment layout ----------------
// Wp layout: [nt][kt][lane][e] with 16 bf16 per lane; within a 16x32 (NxK) tile,
// lane = (n&15) + 16*khalf ; e<8 -> k = kt*32 + khalf*8 + e ; e>=8 -> +16.
__global__ void pack_w_kernel(const float* __restrict__ W, int Nreal, int Kreal,
                              int NT, int KT, unsigned short* __restrict__ Wp) {
  size_t total = (size_t)NT * KT * 512;
  for (size_t i = (size_t)blockIdx.x * 256 + threadIdx.x; i < total;
       i += (size_t)gridDim.x * 256) {
    int e    = (int)(i & 15);
    int lane = (int)((i >> 4) & 31);
    size_t tile = i >> 9;
    int kt = (int)(tile % KT);
    int nt = (int)(tile / KT);
    int n  = nt * 16 + (lane & 15);
    int kh = lane >> 4;
    int kl = (e < 8) ? (kh * 8 + e) : (16 + kh * 8 + (e - 8));
    int k  = kt * 32 + kl;
    float val = (n < Nreal && k < Kreal) ? W[(size_t)n * Kreal + k] : 0.0f;
    Wp[i] = f2bf(val);
  }
}

// ---------------- f32 -> bf16 convert ----------------
__global__ void cvt_bf16_kernel(const float* __restrict__ in,
                                unsigned short* __restrict__ out, size_t n) {
  size_t i = (size_t)blockIdx.x * 256 + threadIdx.x;
  if (i < n) out[i] = f2bf(in[i]);
}

// ---------------- fused 12-way LSTM, one feature x 16 rows per block ----------------
// h state ping-pongs between two LDS buffers; one barrier per step.
__global__ void lstm_kernel(const float* __restrict__ temporal,   // [B,T,F]
                            const float* __restrict__ tmask,      // [B,T,F]
                            const float* __restrict__ h0,         // [F,B,H]
                            const float* __restrict__ c0,         // [F,B,H]
                            const float* __restrict__ W_ih,       // [F,4H,2]
                            const float* __restrict__ b_ih,       // [F,4H]
                            const float* __restrict__ b_hh,       // [F,4H]
                            const unsigned short* __restrict__ Wp, // [F][128][16][32][16]
                            unsigned short* __restrict__ z)        // bf16 [B, ZK]
{
  extern __shared__ unsigned short hbuf[];   // [2][16][HSTR]
  const int f    = blockIdx.y;
  const int b0   = blockIdx.x * 16;
  const int tid  = threadIdx.x;
  const int wave = tid >> 5;
  const int lane = tid & 31;
  const int ln   = lane & 15;
  const int kh   = lane >> 4;

  // stage h0 as bf16 into buffer 0
  for (int i = tid; i < 16 * Hn; i += 256) {
    int r = i >> 9, cidx = i & (Hn - 1);
    hbuf[r * HSTR + cidx] = f2bf(h0[((size_t)f * Bn + b0 + r) * Hn + cidx]);
  }

  // persistent c state in registers; wave owns hidden tiles wave+8j
  float creg[4][8];
  for (int j = 0; j < 4; ++j) {
    int nh = (wave + 8 * j) * 16 + ln;
    for (int v = 0; v < 8; ++v) {
      int m = v + 8 * kh;
      creg[j][v] = c0[((size_t)f * Bn + b0 + m) * Hn + nh];
    }
  }

  const unsigned short* wpf = Wp + (size_t)f * (128 * 16 * 512);

#pragma unroll 1
  for (int t = 0; t < Tn; ++t) {
    __syncthreads();   // all writes of step t-1 (and reads of t-2's buffer) done
    const int p = t & 1;
    const unsigned short* rbuf = hbuf + p * (16 * HSTR);
    unsigned short*       wbuf = hbuf + (p ^ 1) * (16 * HSTR);
#pragma unroll 1
    for (int j = 0; j < 4; ++j) {
      const int nth = wave + 8 * j;
      // single base pointers; every load below is an immediate-offset form
      const unsigned short* pa = rbuf + (ln * HSTR + kh * 8);
      const unsigned short* pb = wpf + (((size_t)nth * 16) * 32 + lane) * 16;

      v8f acc[4] = {};   // i, f, g, o accumulators (16x16 C tiles)
      Frag afr[2];
      {
        const uint4* qa = reinterpret_cast<const uint4*>(pa);
        afr[0].u[0] = qa[0];
        afr[0].u[1] = qa[2];
      }
      Frag bcur;
      {
        const uint4* qb = reinterpret_cast<const uint4*>(pb);
        bcur.u[0] = qb[0];
        bcur.u[1] = qb[1];
      }
      // flat (kt,g) tile loop, one-fragment lookahead on B, double-buffered A
#pragma unroll
      for (int it = 0; it < 64; ++it) {
        const int kt = it >> 2, g = it & 3;
        Frag bnxt;
        if (it < 63) {
          const int itn = it + 1, ktn = itn >> 2, gn = itn & 3;
          const uint4* qb = reinterpret_cast<const uint4*>(pb + ktn * 512 + gn * GSTRIDE);
          bnxt.u[0] = qb[0];
          bnxt.u[1] = qb[1];
        }
        if (g == 0 && kt < 15) {   // prefetch next A slab while this kt's gates run
          const uint4* qa = reinterpret_cast<const uint4*>(pa + (kt + 1) * 32);
          afr[(kt + 1) & 1].u[0] = qa[0];
          afr[(kt + 1) & 1].u[1] = qa[2];
        }
        acc[g] = __builtin_amdgcn_wmma_f32_16x16x32_bf16(
            false, afr[kt & 1].v, false, bcur.v, (short)0, acc[g], false, false);
        if (it < 63) bcur = bnxt;
      }

      // pointwise gates; per-gate constants reloaded (L2-resident, coalesced)
      const int nh = nth * 16 + ln;
      float u0[4], u1[4], ub[4];
      for (int g = 0; g < 4; ++g) {
        int col = g * Hn + nh;
        const float* wp = W_ih + ((size_t)f * GH + col) * 2;
        u0[g] = wp[0];
        u1[g] = wp[1];
        ub[g] = b_ih[(size_t)f * GH + col] + b_hh[(size_t)f * GH + col];
      }
      for (int v = 0; v < 8; ++v) {
        int m = v + 8 * kh;
        size_t bi = ((size_t)(b0 + m) * Tn + t) * Fn + f;
        float val = temporal[bi];
        float msk = tmask[bi];
        float gi = acc[0][v] + val * u0[0] + msk * u1[0] + ub[0];
        float gf = acc[1][v] + val * u0[1] + msk * u1[1] + ub[1];
        float gg = acc[2][v] + val * u0[2] + msk * u1[2] + ub[2];
        float go = acc[3][v] + val * u0[3] + msk * u1[3] + ub[3];
        float cn = sigf(gf) * creg[j][v] + sigf(gi) * tanhf(gg);
        creg[j][v] = cn;
        float hv = sigf(go) * tanhf(cn);
        unsigned short hb = f2bf(hv);
        wbuf[m * HSTR + nh] = hb;
        if (t == Tn - 1)
          z[(size_t)(b0 + m) * ZK + f * Hn + nh] = hb;   // cat_temp
      }
    }
  }
}

// ---------------- generic WMMA GEMM: out = relu(A @ Wp^T + bias) ----------------
// A panel chunks are staged into LDS with async global->LDS copies; all waves
// cooperate in staging, inactive waves skip only the matrix/epilogue work.
__global__ void gemm_kernel(const unsigned short* __restrict__ A, int lda,  // bf16
                            const unsigned short* __restrict__ Wp, int KT,
                            const float* __restrict__ bias, int Nreal, int Npad,
                            float* __restrict__ outF, int ldF,
                            unsigned short* __restrict__ outB, int ldB, int colofs)
{
  extern __shared__ unsigned short atile[];   // [16][HSTR]
  const int m0   = blockIdx.x * 16;
  const int tid  = threadIdx.x;
  const int wave = tid >> 5;
  const int lane = tid & 31;
  const int ln   = lane & 15, kh = lane >> 4;
  const int nt   = blockIdx.y * 8 + wave;
  const bool active = (nt * 16) < Npad;

  v8f acc = {};
  const int nchunk = (KT + 15) >> 4;
#pragma unroll 1
  for (int kc = 0; kc < nchunk; ++kc) {
    const int ktlo = kc << 4;
    const int KC = (KT - ktlo < 16) ? (KT - ktlo) : 16;
    // ---- async stage A[16][KC*32] into LDS (16-byte pieces) ----
    const int npieces = 16 * KC * 4;
    for (int i = tid; i < npieces; i += 256) {
      const int row = i / (KC * 4);
      const int c8  = i - row * (KC * 4);
      unsigned lofs = (unsigned)(size_t)&atile[row * HSTR + c8 * 8];
      unsigned long long ga = (unsigned long long)(size_t)(
          A + (size_t)(m0 + row) * lda + ktlo * 32 + c8 * 8);
      asm volatile("global_load_async_to_lds_b128 %0, %1, off"
                   :: "v"(lofs), "v"(ga) : "memory");
    }
    asm volatile("s_wait_asynccnt 0x0" ::: "memory");
    __syncthreads();

    if (active) {
      const unsigned short* pa = atile + (ln * HSTR + kh * 8);
      const unsigned short* pw = Wp + ((size_t)(nt * KT + ktlo) * 32 + lane) * 16;
      Frag bcur;
      {
        const uint4* qb = reinterpret_cast<const uint4*>(pw);
        bcur.u[0] = qb[0];
        bcur.u[1] = qb[1];
      }
#pragma unroll 4
      for (int kt2 = 0; kt2 < KC; ++kt2) {
        Frag a, bnxt;
        if (kt2 + 1 < KC) {   // prefetch next B fragment under the WMMA
          const uint4* qb = reinterpret_cast<const uint4*>(pw + (kt2 + 1) * 512);
          bnxt.u[0] = qb[0];
          bnxt.u[1] = qb[1];
        }
        const uint4* qa = reinterpret_cast<const uint4*>(pa + kt2 * 32);
        a.u[0] = qa[0];
        a.u[1] = qa[2];
        acc = __builtin_amdgcn_wmma_f32_16x16x32_bf16(
            false, a.v, false, bcur.v, (short)0, acc, false, false);
        if (kt2 + 1 < KC) bcur = bnxt;
      }
    }
    __syncthreads();   // LDS consumed before next chunk overwrites
  }

  if (active) {
    for (int v = 0; v < 8; ++v) {
      int m = m0 + v + 8 * kh;
      int n = nt * 16 + ln;
      if (n < Nreal) {
        float x = fmaxf(acc[v] + bias[n], 0.0f);
        if (outF) outF[(size_t)m * ldF + n] = x;
        if (outB) outB[(size_t)m * ldB + colofs + n] = f2bf(x);
      }
    }
  }
}

// ---------------- BCE-with-logits mean loss ----------------
__global__ void loss_kernel(const float* __restrict__ pred,
                            const float* __restrict__ tgt,
                            float* __restrict__ out) {
  __shared__ float red[256];
  float s = 0.0f;
  for (int i = threadIdx.x; i < Bn * TGT; i += 256) {
    float p = pred[i], y = tgt[i];
    s += fmaxf(p, 0.0f) - p * y + log1pf(__expf(-fabsf(p)));
  }
  red[threadIdx.x] = s;
  __syncthreads();
  for (int off = 128; off > 0; off >>= 1) {
    if (threadIdx.x < off) red[threadIdx.x] += red[threadIdx.x + off];
    __syncthreads();
  }
  if (threadIdx.x == 0) out[0] = red[0] / (float)(Bn * TGT);
}

// ---------------- host-side orchestration ----------------
static inline size_t wsalloc(size_t& off, size_t bytes) {
  size_t o = off;
  off += (bytes + 255) & ~(size_t)255;
  return o;
}
static inline void launch_pack(const float* W, int Nreal, int Kreal, int NT, int KT,
                               unsigned short* Wp, hipStream_t s) {
  size_t total = (size_t)NT * KT * 512;
  int blocks = (int)((total + 255) / 256);
  pack_w_kernel<<<blocks, 256, 0, s>>>(W, Nreal, Kreal, NT, KT, Wp);
}

extern "C" void kernel_launch(void* const* d_in, const int* in_sizes, int n_in,
                              void* d_out, int out_size, void* d_ws, size_t ws_size,
                              hipStream_t stream) {
  (void)in_sizes; (void)n_in; (void)out_size; (void)ws_size;
  const float* statf    = (const float*)d_in[0];
  const float* temporal = (const float*)d_in[1];
  const float* tmask    = (const float*)d_in[2];
  const float* targets  = (const float*)d_in[3];
  const float* h0       = (const float*)d_in[4];
  const float* c0       = (const float*)d_in[5];
  const float* W_ih     = (const float*)d_in[6];
  const float* W_hh     = (const float*)d_in[7];
  const float* b_ih     = (const float*)d_in[8];
  const float* b_hh     = (const float*)d_in[9];
  const float* sW1 = (const float*)d_in[10]; const float* sb1 = (const float*)d_in[11];
  const float* sW2 = (const float*)d_in[12]; const float* sb2 = (const float*)d_in[13];
  const float* sW3 = (const float*)d_in[14]; const float* sb3 = (const float*)d_in[15];
  const float* dW1 = (const float*)d_in[16]; const float* db1 = (const float*)d_in[17];
  const float* dW2 = (const float*)d_in[18]; const float* db2 = (const float*)d_in[19];
  const float* dW3 = (const float*)d_in[20]; const float* db3 = (const float*)d_in[21];

  char* ws = (char*)d_ws;
  size_t off = 0;
  unsigned short* WpHH = (unsigned short*)(ws + wsalloc(off, (size_t)Fn * 128 * 16 * 512 * 2));
  unsigned short* zbuf = (unsigned short*)(ws + wsalloc(off, (size_t)Bn * ZK * 2));
  unsigned short* sbin = (unsigned short*)(ws + wsalloc(off, (size_t)Bn * SD * 2));
  unsigned short* s1   = (unsigned short*)(ws + wsalloc(off, (size_t)Bn * SE * 2));
  unsigned short* s2   = (unsigned short*)(ws + wsalloc(off, (size_t)Bn * SE * 2));
  unsigned short* m1   = (unsigned short*)(ws + wsalloc(off, (size_t)Bn * MH * 2));
  unsigned short* m2   = (unsigned short*)(ws + wsalloc(off, (size_t)Bn * MH * 2));
  unsigned short* sW1p = (unsigned short*)(ws + wsalloc(off, (size_t)16 * 2 * 512 * 2));
  unsigned short* sW2p = (unsigned short*)(ws + wsalloc(off, (size_t)16 * 8 * 512 * 2));
  unsigned short* sW3p = (unsigned short*)(ws + wsalloc(off, (size_t)16 * 8 * 512 * 2));
  unsigned short* dW1p = (unsigned short*)(ws + wsalloc(off, (size_t)64 * 200 * 512 * 2));
  unsigned short* dW2p = (unsigned short*)(ws + wsalloc(off, (size_t)64 * 32 * 512 * 2));
  unsigned short* dW3p = (unsigned short*)(ws + wsalloc(off, (size_t)1 * 32 * 512 * 2));

  // pack all weights into WMMA fragment layout (bf16)
  for (int f = 0; f < Fn; ++f)
    launch_pack(W_hh + (size_t)f * GH * Hn, GH, Hn, 128, 16,
                WpHH + (size_t)f * 128 * 16 * 512, stream);
  launch_pack(sW1, SE, SD, 16, 2, sW1p, stream);
  launch_pack(sW2, SE, SE, 16, 8, sW2p, stream);
  launch_pack(sW3, SE, SE, 16, 8, sW3p, stream);
  launch_pack(dW1, MH, ZK, 64, 200, dW1p, stream);
  launch_pack(dW2, MH, MH, 64, 32, dW2p, stream);
  launch_pack(dW3, TGT, MH, 1, 32, dW3p, stream);

  // static features -> bf16
  {
    size_t n = (size_t)Bn * SD;
    cvt_bf16_kernel<<<(int)((n + 255) / 256), 256, 0, stream>>>(statf, sbin, n);
  }

  const size_t smem1 = (size_t)16 * HSTR * 2;       // gemm staging buffer
  const size_t smem2 = (size_t)2 * 16 * HSTR * 2;   // lstm ping-pong buffers

  // fused LSTM bank -> z[:, 0:6144]
  lstm_kernel<<<dim3(Bn / 16, Fn), 256, smem2, stream>>>(
      temporal, tmask, h0, c0, W_ih, b_ih, b_hh, WpHH, zbuf);

  // static MLP chain -> z[:, 6144:6400]
  gemm_kernel<<<dim3(Bn / 16, 2), 256, smem1, stream>>>(sbin, SD, sW1p, 2, sb1, SE, SE,
                                                        nullptr, 0, s1, SE, 0);
  gemm_kernel<<<dim3(Bn / 16, 2), 256, smem1, stream>>>(s1, SE, sW2p, 8, sb2, SE, SE,
                                                        nullptr, 0, s2, SE, 0);
  gemm_kernel<<<dim3(Bn / 16, 2), 256, smem1, stream>>>(s2, SE, sW3p, 8, sb3, SE, SE,
                                                        nullptr, 0, zbuf, ZK, Fn * Hn);

  // head MLP
  gemm_kernel<<<dim3(Bn / 16, 8), 256, smem1, stream>>>(zbuf, ZK, dW1p, 200, db1, MH, MH,
                                                        nullptr, 0, m1, MH, 0);
  gemm_kernel<<<dim3(Bn / 16, 8), 256, smem1, stream>>>(m1, MH, dW2p, 32, db2, MH, MH,
                                                        nullptr, 0, m2, MH, 0);
  gemm_kernel<<<dim3(Bn / 16, 1), 256, smem1, stream>>>(m2, MH, dW3p, 32, db3, TGT, 16,
                                                        (float*)d_out, TGT, nullptr, 0, 0);

  // BCE loss -> d_out[8192]
  loss_kernel<<<1, 256, 0, stream>>>((const float*)d_out, targets,
                                     (float*)d_out + (size_t)Bn * TGT);
}
